// SelfAttn1d2_11020886082283
// MI455X (gfx1250) — matrix operand order repaired and checked by reading, not compiled, not run on hardware
//
#include <hip/hip_runtime.h>
#include <stdint.h>

#define BATCH 8
#define CH    512
#define OCH   64
#define NSEQ  2048

typedef __attribute__((ext_vector_type(16))) __bf16 v16bf;
typedef __attribute__((ext_vector_type(2)))  __bf16 v2bf;
typedef __attribute__((ext_vector_type(8)))  float  v8f;

union FragAB { v16bf v; uint32_t u[8]; uint4 q[2]; };
union FragC  { v8f   v; float    f[8]; };

// fp32 -> bf16 using the native converts (v_cvt_pk_bf16_f32 on gfx1250)
__device__ __forceinline__ uint32_t pack2(float a, float b) {
  v2bf t;
  t[0] = (__bf16)a;
  t[1] = (__bf16)b;
  return __builtin_bit_cast(uint32_t, t);
}
__device__ __forceinline__ uint16_t f2bf(float x) {
  __bf16 t = (__bf16)x;
  return __builtin_bit_cast(uint16_t, t);
}

// ---- CDNA5 async memory->LDS copy (ASYNCcnt-tracked), ISA §15.18.3 op 98 ----
__device__ __forceinline__ uint32_t lds_offset_of(const void* p) {
  return (uint32_t)(uintptr_t)(__attribute__((address_space(3))) const void*)p;
}
__device__ __forceinline__ void async_load_b128(uint32_t lds_off, const void* gptr) {
  asm volatile("global_load_async_to_lds_b128 %0, %1, off"
               :: "v"(lds_off), "v"((uint64_t)(uintptr_t)gptr)
               : "memory");
}
__device__ __forceinline__ void wait_asynccnt0() {
  asm volatile("s_wait_asynccnt 0" ::: "memory");
}

// ---------------------------------------------------------------------------
// Projection GEMM: Y[oc, n] = W[oc, :] . x[:, n] + bias[oc]   (per batch)
// Each wave computes a 16(oc) x 64(n) tile with v_wmma_f32_16x16x32_bf16.
// transposed=1 -> store Y as out[b][n][OC] (bf16)  (used for f and g)
// transposed=0 -> store Y as out[b][oc][N] (bf16)  (used for h)
// ---------------------------------------------------------------------------
__global__ __launch_bounds__(128)
void proj_gemm_kernel(const float* __restrict__ W, const float* __restrict__ bias,
                      const float* __restrict__ x, uint16_t* __restrict__ out,
                      int OC, int transposed)
{
  const int lane = threadIdx.x & 31;
  const int wave = threadIdx.x >> 5;
  const int half = lane >> 4;
  const int lm   = lane & 15;
  const int b    = blockIdx.z;
  const int ocb  = (blockIdx.y * 4 + wave) * 16;
  const int nb   = blockIdx.x * 64;

  const float* xb = x + (size_t)b * CH * NSEQ;

  FragC acc[4];
#pragma unroll
  for (int t = 0; t < 4; ++t)
#pragma unroll
    for (int r = 0; r < 8; ++r) acc[t].f[r] = 0.0f;

  const float* wrow = W + (size_t)(ocb + lm) * CH;

  for (int kb = 0; kb < CH; kb += 32) {
    // A fragment: 16x32 bf16, M = lm, K mapping per CDNA5 ISA 16-bit A layout
    FragAB a;
#pragma unroll
    for (int v = 0; v < 4; ++v) {
      int k0 = kb + 8 * half + 2 * v;
      a.u[v] = pack2(wrow[k0], wrow[k0 + 1]);
      int k1 = kb + 16 + 8 * half + 2 * v;
      a.u[4 + v] = pack2(wrow[k1], wrow[k1 + 1]);
    }
#pragma unroll
    for (int t = 0; t < 4; ++t) {
      const int n = nb + t * 16 + lm;
      // B fragment: 32x16 bf16, N = lm, K = 8*half + 2v (+8 for v>=4)
      FragAB bb;
#pragma unroll
      for (int v = 0; v < 8; ++v) {
        int k = kb + 8 * half + 2 * v + (v >= 4 ? 8 : 0);
        bb.u[v] = pack2(xb[(size_t)k * NSEQ + n], xb[(size_t)(k + 1) * NSEQ + n]);
      }
      acc[t].v = __builtin_amdgcn_wmma_f32_16x16x32_bf16(
          false, a.v, false, bb.v, (short)0, acc[t].v, false, false);
    }
  }

  float bv[8];
#pragma unroll
  for (int r = 0; r < 8; ++r) bv[r] = bias[ocb + r + 8 * half];

  if (transposed) {
    // out[b][n][OC]: lane holds 8 consecutive oc -> one 16B store
#pragma unroll
    for (int t = 0; t < 4; ++t) {
      const int n = nb + t * 16 + lm;
      uint32_t p0 = pack2(acc[t].f[0] + bv[0], acc[t].f[1] + bv[1]);
      uint32_t p1 = pack2(acc[t].f[2] + bv[2], acc[t].f[3] + bv[3]);
      uint32_t p2 = pack2(acc[t].f[4] + bv[4], acc[t].f[5] + bv[5]);
      uint32_t p3 = pack2(acc[t].f[6] + bv[6], acc[t].f[7] + bv[7]);
      uint16_t* dst = out + ((size_t)b * NSEQ + n) * (size_t)OC + ocb + 8 * half;
      *(uint4*)dst = make_uint4(p0, p1, p2, p3);
    }
  } else {
    // out[b][oc][N]
#pragma unroll
    for (int t = 0; t < 4; ++t) {
      const int n = nb + t * 16 + lm;
#pragma unroll
      for (int r = 0; r < 8; ++r) {
        const int c = ocb + r + 8 * half;
        out[((size_t)b * OC + c) * NSEQ + n] = f2bf(acc[t].f[r] + bv[r]);
      }
    }
  }
}

// ---------------------------------------------------------------------------
// Fused flash-attention kernel (register-resident fragments, minimal LDS).
//   fT [b][n][64] bf16 (keys), gT [b][m][64] bf16 (queries), hM [b][c][N] bf16
//   One workgroup (8 waves) per (batch, 32-wide m-block).
//   S[n,m] = f^T g, online softmax over n, acc[c,m] += h P.
//   f/h A-fragments are two contiguous 16B runs per lane -> direct
//   global_load_b128, issued at iteration top to overlap S+softmax phases.
//   g tile staged once via GLOBAL_LOAD_ASYNC_TO_LDS_B128.
// ---------------------------------------------------------------------------
__global__ __launch_bounds__(256)
void flash_attn_kernel(const uint16_t* __restrict__ fT,
                       const uint16_t* __restrict__ gT,
                       const uint16_t* __restrict__ hM,
                       const float* __restrict__ x,
                       const float* __restrict__ gamma_p,
                       float* __restrict__ out)
{
  __shared__ __align__(16) uint16_t g_lds[32][64];   // [m_local][o]   4 KB
  __shared__ float    S_lds[32][32];                 // [n_local][m]   4 KB
  __shared__ __align__(16) uint16_t Pt_lds[32][32];  // [m][n_local]   2 KB
  __shared__ float scale_lds[32];
  __shared__ float Mrun[32];
  __shared__ float Lrun[32];

  const int tid  = threadIdx.x;
  const int lane = tid & 31;
  const int wave = tid >> 5;
  const int half = lane >> 4;
  const int lm   = lane & 15;
  const int b    = blockIdx.y;
  const int mb   = blockIdx.x * 32;

  const uint16_t* fb = fT + (size_t)b * NSEQ * OCH;
  const uint16_t* hb = hM + (size_t)b * CH * NSEQ;

  // stage g tile once via async copy: 32 rows x 8 16B-chunks
  {
    const int grow = tid >> 3, gchunk = tid & 7;
    async_load_b128(lds_offset_of(&g_lds[grow][gchunk * 8]),
                    gT + ((size_t)b * NSEQ + mb + grow) * OCH + gchunk * 8);
  }
  if (tid < 32) { Mrun[tid] = -3.0e38f; Lrun[tid] = 0.0f; }
  wait_asynccnt0();
  __syncthreads();

  // g B-fragments (waves 0/1 only): [msub][khalf], 32x16 bf16 each
  FragAB gf[2][2];
  if (wave < 2) {
#pragma unroll
    for (int msub = 0; msub < 2; ++msub)
#pragma unroll
      for (int kh = 0; kh < 2; ++kh)
#pragma unroll
        for (int v = 0; v < 8; ++v)
          gf[msub][kh].u[v] =
              *(const uint32_t*)&g_lds[msub * 16 + lm][kh * 32 + 8 * half + 2 * v + (v >= 4 ? 8 : 0)];
  }

  FragC acc[4][2];
#pragma unroll
  for (int ct = 0; ct < 4; ++ct)
#pragma unroll
    for (int ms = 0; ms < 2; ++ms)
#pragma unroll
      for (int r = 0; r < 8; ++r) acc[ct][ms].f[r] = 0.0f;

  for (int ns = 0; ns < NSEQ / 32; ++ns) {
    const int nb = ns * 32;

    // issue h A-fragment loads now; consumed after two barriers, so their
    // latency overlaps the S-WMMA and softmax phases (LOADcnt pends freely).
    FragAB hf[4];
#pragma unroll
    for (int ct = 0; ct < 4; ++ct) {
      const uint16_t* hrow = hb + (size_t)(wave * 64 + ct * 16 + lm) * NSEQ + nb;
      hf[ct].q[0] = *(const uint4*)(hrow + 8 * half);
      hf[ct].q[1] = *(const uint4*)(hrow + 16 + 8 * half);
    }

    // S tile: waves 0/1 each compute a 16(n) x 32(m) slab, A direct from global
    if (wave < 2) {
      const int nsub = wave;
      const uint16_t* frow = fb + (size_t)(nb + nsub * 16 + lm) * OCH;
      FragAB af[2];
#pragma unroll
      for (int kh = 0; kh < 2; ++kh) {
        af[kh].q[0] = *(const uint4*)(frow + kh * 32 + 8 * half);
        af[kh].q[1] = *(const uint4*)(frow + kh * 32 + 16 + 8 * half);
      }
#pragma unroll
      for (int msub = 0; msub < 2; ++msub) {
        FragC s;
#pragma unroll
        for (int r = 0; r < 8; ++r) s.f[r] = 0.0f;
        s.v = __builtin_amdgcn_wmma_f32_16x16x32_bf16(
            false, af[0].v, false, gf[msub][0].v, (short)0, s.v, false, false);
        s.v = __builtin_amdgcn_wmma_f32_16x16x32_bf16(
            false, af[1].v, false, gf[msub][1].v, (short)0, s.v, false, false);
#pragma unroll
        for (int r = 0; r < 8; ++r)
          S_lds[nsub * 16 + r + 8 * half][msub * 16 + lm] = s.f[r];
      }
    }
    __syncthreads();

    // online softmax over n (one thread per m column; wave 0)
    if (tid < 32) {
      const int t = tid;
      float mold = Mrun[t];
      float mloc = -3.0e38f;
      for (int n = 0; n < 32; ++n) mloc = fmaxf(mloc, S_lds[n][t]);
      const float mnew = fmaxf(mold, mloc);
      const float a = __expf(mold - mnew);
      float sum = 0.0f;
      for (int n = 0; n < 32; ++n) {
        const float e = __expf(S_lds[n][t] - mnew);
        Pt_lds[t][n] = f2bf(e);
        sum += e;
      }
      Lrun[t] = Lrun[t] * a + sum;
      Mrun[t] = mnew;
      scale_lds[t] = a;
    }
    __syncthreads();

    // all 8 waves: rescale accumulators, acc += h_frag x P_frag
    FragAB pf[2];
#pragma unroll
    for (int msub = 0; msub < 2; ++msub)
#pragma unroll
      for (int v = 0; v < 8; ++v)
        pf[msub].u[v] =
            *(const uint32_t*)&Pt_lds[msub * 16 + lm][8 * half + 2 * v + (v >= 4 ? 8 : 0)];

    const float s0 = scale_lds[lm];
    const float s1 = scale_lds[16 + lm];
#pragma unroll
    for (int ct = 0; ct < 4; ++ct) {
#pragma unroll
      for (int r = 0; r < 8; ++r) { acc[ct][0].f[r] *= s0; acc[ct][1].f[r] *= s1; }
      acc[ct][0].v = __builtin_amdgcn_wmma_f32_16x16x32_bf16(
          false, hf[ct].v, false, pf[0].v, (short)0, acc[ct][0].v, false, false);
      acc[ct][1].v = __builtin_amdgcn_wmma_f32_16x16x32_bf16(
          false, hf[ct].v, false, pf[1].v, (short)0, acc[ct][1].v, false, false);
    }
    // no trailing barrier needed: next iteration's post-S barrier orders all
    // cross-wave LDS reuse (Pt/scale are written only after it; S reads end before the post-softmax barrier).
  }
  __syncthreads();   // make final Lrun visible to all waves

  // epilogue: out = gamma * (acc / L) + x
  const float gma = gamma_p[0];
#pragma unroll
  for (int msub = 0; msub < 2; ++msub) {
    const float linv = 1.0f / Lrun[msub * 16 + lm];
    const int m = mb + msub * 16 + lm;
#pragma unroll
    for (int ct = 0; ct < 4; ++ct) {
#pragma unroll
      for (int r = 0; r < 8; ++r) {
        const int c = wave * 64 + ct * 16 + r + 8 * half;
        const size_t off = ((size_t)b * CH + c) * NSEQ + m;
        out[off] = gma * (acc[ct][msub].f[r] * linv) + x[off];
      }
    }
  }
}

// ---------------------------------------------------------------------------
extern "C" void kernel_launch(void* const* d_in, const int* in_sizes, int n_in,
                              void* d_out, int out_size, void* d_ws, size_t ws_size,
                              hipStream_t stream) {
  const float* x     = (const float*)d_in[0];
  const float* wf    = (const float*)d_in[1];
  const float* bf    = (const float*)d_in[2];
  const float* wg    = (const float*)d_in[3];
  const float* bg    = (const float*)d_in[4];
  const float* wh    = (const float*)d_in[5];
  const float* bh    = (const float*)d_in[6];
  const float* gamma = (const float*)d_in[7];
  float* out = (float*)d_out;

  // workspace: f (2 MB) + g (2 MB) + h (16 MB), all bf16
  uint16_t* fT = (uint16_t*)d_ws;
  uint16_t* gT = fT + (size_t)BATCH * NSEQ * OCH;
  uint16_t* hM = gT + (size_t)BATCH * NSEQ * OCH;

  const dim3 blk(128);
  proj_gemm_kernel<<<dim3(NSEQ / 64, OCH / 64, BATCH), blk, 0, stream>>>(wf, bf, x, fT, OCH, 1);
  proj_gemm_kernel<<<dim3(NSEQ / 64, OCH / 64, BATCH), blk, 0, stream>>>(wg, bg, x, gT, OCH, 1);
  proj_gemm_kernel<<<dim3(NSEQ / 64, CH / 64,  BATCH), blk, 0, stream>>>(wh, bh, x, hM, CH, 0);

  flash_attn_kernel<<<dim3(NSEQ / 32, BATCH), dim3(256), 0, stream>>>(fT, gT, hM, x, gamma, out);
}
